// Int4PackedLinear_46119358825249
// MI455X (gfx1250) — matrix operand order repaired and checked by reading, not compiled, not run on hardware
//
#include <hip/hip_runtime.h>
#include <hip/hip_bf16.h>
#include <stdint.h>

// ---------------------------------------------------------------------------
// y[m,n] = sum_k x[m,k] * w[n,k] + bias[n]
//   w[n,k] = bf16( (reg_a[n,k]*low4_scale + reg_b[n,k]) * scale )
// M = 128, K = 4096, N = 11008.
//
// Fast path (needs 1MB workspace):
//   pre-pass: x bf16 -> f16 (exact in f16 normal range) into d_ws
//   main:     w' = f16(a*s1 + b) via magic-number byte->f16:
//               u    = byte ^ 0x80          (a+128; 1 xor per 4 weights)
//               pk16 = v_perm(0x64646464)   (0x6400|u = 1024+u, exact)
//               af   = pk16 - 1152          (exact; |a|<=7)
//               w'   = pk_fma(af, s1, bf)   (f16; tighter than ref's bf16 w)
//             y = s2 * (x @ w'.T) + bias  via v_wmma_f32_16x16x32_f16,
//             s2/bias folded into epilogue v_fma_mix.
//   schedule: all 8 A fragments preloaded per K-chunk BEFORE the dequant
//             VALU block, so load latency hides under pk-math and the 8
//             WMMAs issue back-to-back (avoid per-WMMA s_wait_loadcnt 0).
// Tiling: wave -> one 16-col N tile x all eight 16-row M tiles
//         (one dequantized B fragment feeds 8 WMMAs); 172 blocks x 4 waves.
// Weights stream once per call (~90MB -> ~4us @ 23.3TB/s HBM; L2-resident
// across graph replays -> keep default temporal hints).
// ---------------------------------------------------------------------------

typedef _Float16 h2 __attribute__((ext_vector_type(2)));
typedef __attribute__((ext_vector_type(16))) _Float16 v16h;
typedef __attribute__((ext_vector_type(16))) __bf16   v16bf;
typedef __attribute__((ext_vector_type(8)))  float    v8f;

#define IN_F  4096
#define OUT_F 11008
#define M_TOT 128

// ---------------- pre-pass: x bf16 -> f16 into workspace --------------------
__global__ __launch_bounds__(256)
void x_bf16_to_f16(const uint16_t* __restrict__ xin, uint16_t* __restrict__ xout) {
    const int i = (blockIdx.x * 256 + threadIdx.x) * 8;   // 8 elements/thread
    uint4 v = *reinterpret_cast<const uint4*>(xin + i);
    uint32_t wv[4] = { v.x, v.y, v.z, v.w };
    uint32_t r[4];
#pragma unroll
    for (int j = 0; j < 4; ++j) {
        float f0 = __builtin_bit_cast(float, wv[j] << 16);
        float f1 = __builtin_bit_cast(float, wv[j] & 0xFFFF0000u);
        h2 p = { (_Float16)f0, (_Float16)f1 };
        r[j] = __builtin_bit_cast(uint32_t, p);
    }
    uint4 o = { r[0], r[1], r[2], r[3] };
    *reinterpret_cast<uint4*>(xout + i) = o;
}

// ---------------- fast path: f16 WMMA with magic-number dequant -------------
__global__ __launch_bounds__(128, 2)
void int4_linear_wmma_f16(const _Float16* __restrict__ xh,
                          const int8_t*  __restrict__ ra,
                          const int8_t*  __restrict__ rb,
                          const float*   __restrict__ scale_p,
                          const float*   __restrict__ low4_p,
                          const __hip_bfloat16* __restrict__ bias,
                          __hip_bfloat16* __restrict__ out) {
    const int lane = threadIdx.x & 31;
    const int wave = threadIdx.x >> 5;
    const int row  = lane & 15;
    const int half = lane >> 4;

    const int n = blockIdx.x * 64 + wave * 16 + row;

    const float s2 = scale_p[0];
    const _Float16 s1h = (_Float16)low4_p[0];
    const h2 s1pk = { s1h, s1h };
    const h2 koff = { (_Float16)1152.0f, (_Float16)1152.0f };
    const uint32_t K64 = 0x64646464u;   // f16 exponent byte for 1024+u magic

    v8f acc[8];
#pragma unroll
    for (int t = 0; t < 8; ++t) acc[t] = (v8f){0.f,0.f,0.f,0.f,0.f,0.f,0.f,0.f};

    const int8_t* pa = ra + (size_t)n * IN_F + half * 16;
    const int8_t* pb = rb + (size_t)n * IN_F + half * 16;
    const _Float16* px = xh + (size_t)row * IN_F + half * 16;

    for (int k0 = 0; k0 < IN_F; k0 += 32) {
        // ---- issue ALL loads for this chunk first (weights + 8 A frags) ----
        int4 apk = *reinterpret_cast<const int4*>(pa + k0);
        int4 bpk = *reinterpret_cast<const int4*>(pb + k0);

        uint4 Aq[16];
#pragma unroll
        for (int t = 0; t < 8; ++t) {
            const _Float16* p = px + (size_t)t * 16 * IN_F + k0;
            Aq[2 * t]     = *reinterpret_cast<const uint4*>(p);      // K lo half
            Aq[2 * t + 1] = *reinterpret_cast<const uint4*>(p + 8);  // K hi half
        }

        __builtin_prefetch(pa + k0 + 256, 0, 1);
        __builtin_prefetch(pb + k0 + 256, 0, 1);

        // ---- dequant (independent of A loads -> hides their latency) ------
        union { uint32_t u[8]; v16h v; } Bf;
        const uint32_t aw[4] = { (uint32_t)apk.x, (uint32_t)apk.y,
                                 (uint32_t)apk.z, (uint32_t)apk.w };
        const uint32_t bw[4] = { (uint32_t)bpk.x, (uint32_t)bpk.y,
                                 (uint32_t)bpk.z, (uint32_t)bpk.w };
#pragma unroll
        for (int d = 0; d < 4; ++d) {
            const uint32_t ea = aw[d] ^ 0x80808080u;
            const uint32_t eb = bw[d] ^ 0x80808080u;
            // packed {0x6400|u_hi, 0x6400|u_lo} = {1024+u1, 1024+u0}
            const uint32_t Au01 = __builtin_amdgcn_perm(K64, ea, 0x04010400u);
            const uint32_t Au23 = __builtin_amdgcn_perm(K64, ea, 0x04030402u);
            const uint32_t Bu01 = __builtin_amdgcn_perm(K64, eb, 0x04010400u);
            const uint32_t Bu23 = __builtin_amdgcn_perm(K64, eb, 0x04030402u);
            // subtract offset BEFORE the fma (exact small ints; avoids ulp=16)
            const h2 a01 = __builtin_bit_cast(h2, Au01) - koff;
            const h2 a23 = __builtin_bit_cast(h2, Au23) - koff;
            const h2 b01 = __builtin_bit_cast(h2, Bu01) - koff;
            const h2 b23 = __builtin_bit_cast(h2, Bu23) - koff;
            const h2 w01 = a01 * s1pk + b01;   // v_pk_fma_f16
            const h2 w23 = a23 * s1pk + b23;
            Bf.u[2 * d]     = __builtin_bit_cast(uint32_t, w01);
            Bf.u[2 * d + 1] = __builtin_bit_cast(uint32_t, w23);
        }

        // ---- 8 back-to-back WMMAs, one shared B fragment -------------------
#pragma unroll
        for (int t = 0; t < 8; ++t) {
            union { uint4 q[2]; v16h v; } A;
            A.q[0] = Aq[2 * t];
            A.q[1] = Aq[2 * t + 1];
            acc[t] = __builtin_amdgcn_wmma_f32_16x16x32_f16(
                false, A.v, false, Bf.v, (short)0, acc[t], false, false);
        }
    }

    const float bv = __bfloat162float(bias[n]);
#pragma unroll
    for (int t = 0; t < 8; ++t) {
#pragma unroll
        for (int r = 0; r < 8; ++r) {
            const int m = t * 16 + r + half * 8;
            out[(size_t)m * OUT_F + n] = __float2bfloat16(fmaf(acc[t][r], s2, bv));
        }
    }
}

// ---------------- fallback: bf16 WMMA, dequant in f32 -----------------------
__device__ __forceinline__ uint32_t pack2_bf16(float f0, float f1) {
    union { __hip_bfloat16 h[2]; uint32_t u; } cv;
    cv.h[0] = __float2bfloat16(f0);
    cv.h[1] = __float2bfloat16(f1);
    return cv.u;
}

__global__ __launch_bounds__(128, 2)
void int4_linear_wmma_bf16(const __hip_bfloat16* __restrict__ x,
                           const int8_t*  __restrict__ ra,
                           const int8_t*  __restrict__ rb,
                           const float*   __restrict__ scale_p,
                           const float*   __restrict__ low4_p,
                           const __hip_bfloat16* __restrict__ bias,
                           __hip_bfloat16* __restrict__ out) {
    const int lane = threadIdx.x & 31;
    const int wave = threadIdx.x >> 5;
    const int row  = lane & 15;
    const int half = lane >> 4;
    const int n = blockIdx.x * 64 + wave * 16 + row;

    const float s1 = low4_p[0];
    const float s2 = scale_p[0];

    v8f acc[8];
#pragma unroll
    for (int t = 0; t < 8; ++t) acc[t] = (v8f){0.f,0.f,0.f,0.f,0.f,0.f,0.f,0.f};

    const int8_t* pa = ra + (size_t)n * IN_F + half * 16;
    const int8_t* pb = rb + (size_t)n * IN_F + half * 16;
    const __hip_bfloat16* px = x + (size_t)row * IN_F + half * 16;

    for (int k0 = 0; k0 < IN_F; k0 += 32) {
        int4 apk = *reinterpret_cast<const int4*>(pa + k0);
        int4 bpk = *reinterpret_cast<const int4*>(pb + k0);
        __builtin_prefetch(pa + k0 + 256, 0, 1);
        __builtin_prefetch(pb + k0 + 256, 0, 1);

        union { uint32_t u[8]; v16bf v; } B;
        const uint32_t aw[4] = { (uint32_t)apk.x, (uint32_t)apk.y,
                                 (uint32_t)apk.z, (uint32_t)apk.w };
        const uint32_t bw[4] = { (uint32_t)bpk.x, (uint32_t)bpk.y,
                                 (uint32_t)bpk.z, (uint32_t)bpk.w };
#pragma unroll
        for (int d = 0; d < 4; ++d) {
            const uint32_t ua = aw[d], ub = bw[d];
#pragma unroll
            for (int p = 0; p < 2; ++p) {
                float f0 = fmaf((float)(int8_t)(ua >> (16 * p)),     s1,
                                (float)(int8_t)(ub >> (16 * p)));
                float f1 = fmaf((float)(int8_t)(ua >> (16 * p + 8)), s1,
                                (float)(int8_t)(ub >> (16 * p + 8)));
                B.u[2 * d + p] = pack2_bf16(f0, f1);
            }
        }

#pragma unroll
        for (int t = 0; t < 8; ++t) {
            union { uint4 q[2]; v16bf v; } A;
            const __hip_bfloat16* p = px + (size_t)t * 16 * IN_F + k0;
            A.q[0] = *reinterpret_cast<const uint4*>(p);
            A.q[1] = *reinterpret_cast<const uint4*>(p + 8);
            acc[t] = __builtin_amdgcn_wmma_f32_16x16x32_bf16(
                false, A.v, false, B.v, (short)0, acc[t], false, false);
        }
    }

    const float bv = __bfloat162float(bias[n]);
#pragma unroll
    for (int t = 0; t < 8; ++t) {
#pragma unroll
        for (int r = 0; r < 8; ++r) {
            const int m = t * 16 + r + half * 8;
            out[(size_t)m * OUT_F + n] = __float2bfloat16(fmaf(acc[t][r], s2, bv));
        }
    }
}

extern "C" void kernel_launch(void* const* d_in, const int* in_sizes, int n_in,
                              void* d_out, int out_size, void* d_ws, size_t ws_size,
                              hipStream_t stream) {
    const __hip_bfloat16* x     = (const __hip_bfloat16*)d_in[0];
    const int8_t*         ra    = (const int8_t*)d_in[1];
    const int8_t*         rb    = (const int8_t*)d_in[2];
    const float*          scale = (const float*)d_in[3];
    const float*          low4  = (const float*)d_in[4];
    const __hip_bfloat16* bias  = (const __hip_bfloat16*)d_in[5];
    __hip_bfloat16*       out   = (__hip_bfloat16*)d_out;

    const size_t need = (size_t)M_TOT * IN_F * sizeof(uint16_t);  // 1 MB
    dim3 grid(OUT_F / 64, 1);    // 172 blocks x 4 waves

    if (ws_size >= need && d_ws != nullptr) {
        uint16_t* xh = (uint16_t*)d_ws;
        x_bf16_to_f16<<<(M_TOT * IN_F) / (256 * 8), 256, 0, stream>>>(
            (const uint16_t*)x, xh);
        int4_linear_wmma_f16<<<grid, 128, 0, stream>>>(
            (const _Float16*)xh, ra, rb, scale, low4, bias, out);
    } else {
        int4_linear_wmma_bf16<<<grid, 128, 0, stream>>>(
            x, ra, rb, scale, low4, bias, out);
    }
}
